// EnhancedCrossAttention_25451976196563
// MI455X (gfx1250) — matrix operand order repaired and checked by reading, not compiled, not run on hardware
//
#include <hip/hip_runtime.h>
#include <hip/hip_bf16.h>

typedef __attribute__((ext_vector_type(16))) _Float16 v16h;
typedef __attribute__((ext_vector_type(8)))  float    v8f;

namespace {
constexpr int kDim   = 1024;
constexpr int kHeads = 16;
constexpr int kHd    = 64;
constexpr int kBatch = 2;
constexpr int kNq    = 1024;
constexpr int kNkv   = 4096;
constexpr float kScale = 0.125f;  // HD^-0.5 = 1/8
}  // namespace

// ---- async global->LDS (CDNA5 GLOBAL_LOAD_ASYNC_TO_LDS_B128, ASYNCcnt) ----
#if defined(__gfx1250__) &&                                         \
    __has_builtin(__builtin_amdgcn_global_load_async_to_lds_b128) && \
    __has_builtin(__builtin_amdgcn_s_wait_asynccnt)
#define USE_ASYNC_LDS 1
#else
#define USE_ASYNC_LDS 0
#endif

#if USE_ASYNC_LDS
typedef int v4i_gcc __attribute__((vector_size(4 * sizeof(int))));
typedef __attribute__((address_space(1))) v4i_gcc as1_v4i;
typedef __attribute__((address_space(3))) v4i_gcc as3_v4i;
__device__ __forceinline__ void async_ld16(const void* g, void* l) {
  __builtin_amdgcn_global_load_async_to_lds_b128(
      (as1_v4i*)(unsigned long long)g,
      (as3_v4i*)(unsigned)(unsigned long long)l, 0, 0);
}
// 64 rows x 32 halves tile into [64][40] LDS
__device__ __forceinline__ void async_tile_64x32(const _Float16* g, size_t gld,
                                                 _Float16 (*l)[40], int tid) {
#pragma unroll
  for (int it = 0; it < 2; ++it) {
    int i = tid + it * 128;
    int r = i >> 2, c = i & 3;
    async_ld16(g + (size_t)r * gld + c * 8, &l[r][c * 8]);
  }
}
// 64 rows x 64 halves tile into [64][88] LDS
__device__ __forceinline__ void async_tile_64x64(const _Float16* g, size_t gld,
                                                 _Float16 (*l)[88], int tid) {
#pragma unroll
  for (int it = 0; it < 4; ++it) {
    int i = tid + it * 128;
    int r = i >> 3, c = i & 7;
    async_ld16(g + (size_t)r * gld + c * 8, &l[r][c * 8]);
  }
}
#endif

__device__ __forceinline__ v8f wmma16x16x32(v16h a, v16h b, v8f c) {
  return __builtin_amdgcn_wmma_f32_16x16x32_f16(false, a, false, b, (short)0, c,
                                                false, false);
}

// A fragment (16x32 f16) from row-major LDS tile: 32B contiguous per lane.
__device__ __forceinline__ v16h load_a_frag(const _Float16* t, int ld, int m0,
                                            int k0, int lane) {
  const _Float16* row = t + (size_t)(m0 + (lane & 15)) * ld + k0 + (lane >> 4) * 16;
  v16h a;
#pragma unroll
  for (int j = 0; j < 8; ++j) a[j] = row[j];
#pragma unroll
  for (int j = 0; j < 8; ++j) a[8 + j] = row[8 + j];
  return a;
}

// B fragment (32x16) where B[k][n] = t[n*ld + k]: 32B contiguous per lane.
__device__ __forceinline__ v16h load_bt_frag(const _Float16* t, int ld, int k0,
                                             int n0, int lane) {
  const _Float16* row = t + (size_t)(n0 + (lane & 15)) * ld + k0 + (lane >> 4) * 16;
  v16h b;
#pragma unroll
  for (int j = 0; j < 16; ++j) b[j] = row[j];
  return b;
}

// ---------------------------------------------------------------------------
// One-time: fp32 -> f16 with transpose. D[n*1024+k] = (f16)S[k*1024+n].
// ---------------------------------------------------------------------------
__global__ __launch_bounds__(256) void cvt_transpose_kernel(
    const float* __restrict__ S, _Float16* __restrict__ D) {
  __shared__ _Float16 t[32][33];
  const int k0 = blockIdx.x * 32, n0 = blockIdx.y * 32;
#pragma unroll
  for (int it = 0; it < 4; ++it) {
    int i = threadIdx.y + it * 8;
    t[i][threadIdx.x] = (_Float16)S[(size_t)(k0 + i) * kDim + n0 + threadIdx.x];
  }
  __syncthreads();
#pragma unroll
  for (int it = 0; it < 4; ++it) {
    int i = threadIdx.y + it * 8;
    D[(size_t)(n0 + i) * kDim + k0 + threadIdx.x] = t[threadIdx.x][i];
  }
}

// ---------------------------------------------------------------------------
// Q/K/V projection. Wt pre-transposed [n][k]. Double-buffered LDS +
// register/async prefetch pipeline; one barrier per K-step.
// vMode==0: Y[b,h,row,d] (Q,K).  vMode==1: Y[b,h,d,row] (V^T).
// ---------------------------------------------------------------------------
__global__ __launch_bounds__(128) void proj_qkv_kernel(
    const float* __restrict__ X, const _Float16* __restrict__ Wt,
    const float* __restrict__ bias, _Float16* __restrict__ Y, int nShift,
    int vMode) {
  __shared__ _Float16 Xs[2][64][40];
  __shared__ _Float16 Wts[2][64][40];
  const int lane = threadIdx.x & 31;
  const int wave = threadIdx.x >> 5;
  const int m0 = blockIdx.x * 64;
  const int n0 = blockIdx.y * 64;

  float4 xr[2][2];
#if !USE_ASYNC_LDS
  uint4 wr[2];
#endif
  auto load_x = [&](int k0) {
#pragma unroll
    for (int it = 0; it < 2; ++it) {
      int i = threadIdx.x + it * 128;
      int r = i >> 2, c = i & 3;
      const float4* xs = (const float4*)(X + (size_t)(m0 + r) * kDim + k0);
      xr[it][0] = xs[c * 2];
      xr[it][1] = xs[c * 2 + 1];
#if !USE_ASYNC_LDS
      wr[it] = ((const uint4*)(Wt + (size_t)(n0 + r) * kDim + k0))[c];
#endif
    }
  };
  auto store_x = [&](int p) {
#pragma unroll
    for (int it = 0; it < 2; ++it) {
      int i = threadIdx.x + it * 128;
      int r = i >> 2, c = i & 3;
      union { _Float16 h[8]; uint4 v; } pk;
      pk.h[0] = (_Float16)xr[it][0].x;
      pk.h[1] = (_Float16)xr[it][0].y;
      pk.h[2] = (_Float16)xr[it][0].z;
      pk.h[3] = (_Float16)xr[it][0].w;
      pk.h[4] = (_Float16)xr[it][1].x;
      pk.h[5] = (_Float16)xr[it][1].y;
      pk.h[6] = (_Float16)xr[it][1].z;
      pk.h[7] = (_Float16)xr[it][1].w;
      *(uint4*)&Xs[p][r][c * 8] = pk.v;
#if !USE_ASYNC_LDS
      *(uint4*)&Wts[p][r][c * 8] = wr[it];
#endif
    }
  };

  v8f acc[4] = {};
  load_x(0);
#if USE_ASYNC_LDS
  async_tile_64x32(Wt + (size_t)n0 * kDim, kDim, Wts[0], threadIdx.x);
#endif
  int p = 0;
  for (int k0 = 0; k0 < kDim; k0 += 32) {
#if USE_ASYNC_LDS
    __builtin_amdgcn_s_wait_asynccnt(0);
#endif
    store_x(p);
#if !USE_ASYNC_LDS
    if (k0 + 32 < kDim) load_x(k0 + 32);
#endif
    __syncthreads();
#if USE_ASYNC_LDS
    if (k0 + 32 < kDim) {
      async_tile_64x32(Wt + (size_t)n0 * kDim + k0 + 32, kDim, Wts[p ^ 1],
                       threadIdx.x);
      load_x(k0 + 32);
    }
#endif
    v16h a = load_a_frag(&Xs[p][0][0], 40, wave * 16, 0, lane);
#pragma unroll
    for (int t = 0; t < 4; ++t) {
      v16h b = load_bt_frag(&Wts[p][0][0], 40, 0, t * 16, lane);
      acc[t] = wmma16x16x32(a, b, acc[t]);
    }
    p ^= 1;
  }

  const int nMask = (1 << nShift) - 1;
  if (vMode == 0) {
#pragma unroll
    for (int t = 0; t < 4; ++t) {
      int n = n0 + t * 16 + (lane & 15);
      int h = n >> 6, dd = n & 63;
      float bv = bias[n];
#pragma unroll
      for (int r = 0; r < 8; ++r) {
        int m = m0 + wave * 16 + (lane >> 4) * 8 + r;
        int bi = m >> nShift, row = m & nMask;
        Y[(((size_t)(bi * kHeads + h) << nShift) + row) * kHd + dd] =
            (_Float16)(acc[t][r] + bv);
      }
    }
  } else {
    int m = m0 + wave * 16 + (lane >> 4) * 8;
    int bi = m >> nShift, row = m & nMask;
#pragma unroll
    for (int t = 0; t < 4; ++t) {
      int n = n0 + t * 16 + (lane & 15);
      int h = n >> 6, dd = n & 63;
      float bv = bias[n];
      union { _Float16 h8[8]; uint4 v; } pk;
#pragma unroll
      for (int r = 0; r < 8; ++r) pk.h8[r] = (_Float16)(acc[t][r] + bv);
      *(uint4*)&Y[(((size_t)(bi * kHeads + h) * kHd + dd) << nShift) + row] =
          pk.v;
    }
  }
}

// ---------------------------------------------------------------------------
// Flash attention, O^T formulation, double-buffered K/V tiles with
// async/register prefetch. Qs LDS is reused as Ps after Q fragments load.
// ---------------------------------------------------------------------------
__global__ __launch_bounds__(128) void flash_attn_kernel(
    const _Float16* __restrict__ Qh, const _Float16* __restrict__ Kh,
    const _Float16* __restrict__ Vt, const float* __restrict__ objAll,
    _Float16* __restrict__ O) {
  __shared__ _Float16 QPs[64][88];      // Q tile, then per-wave P tiles
  __shared__ _Float16 Ks[2][64][88];    // [kv][d]
  __shared__ _Float16 Vts[2][64][88];   // [d][kv]
  __shared__ float objs[2][64];
  __shared__ float Lalpha[4][16];
  __shared__ float Lsum[4][16];

  const int lane = threadIdx.x & 31;
  const int wave = threadIdx.x >> 5;
  const int x = lane & 15;
  const int g = lane >> 4;
  const int bh = blockIdx.x;
  const int b = bh >> 4;
  const int h = bh & 15;
  const int q0 = blockIdx.y * 64;

  const _Float16* Qbase = Qh + ((size_t)bh * kNq + q0) * kHd;
  const _Float16* Kbase = Kh + (size_t)bh * kNkv * kHd;
  const _Float16* Vtbase = Vt + (size_t)bh * kHd * kNkv;
  const float* objB = objAll + (size_t)b * kNkv;
  _Float16(*PsW)[88] = &QPs[wave * 16];  // this wave's 16x64 P tile

  {  // stage Q tile once
    const uint4* src = (const uint4*)Qbase;
#pragma unroll
    for (int it = 0; it < 4; ++it) {
      int i = threadIdx.x + it * 128;
      int r = i >> 3, c = i & 7;
      *(uint4*)&QPs[r][c * 8] = src[i];
    }
  }
  __syncthreads();
  const v16h qa0 = load_a_frag(&QPs[0][0], 88, wave * 16, 0, lane);
  const v16h qa1 = load_a_frag(&QPs[0][0], 88, wave * 16, 32, lane);
  __syncthreads();  // all waves hold Q frags; QPs may be reused as Ps

  v8f acc[4] = {};  // O^T: tile t -> rows d = t*16+g*8+r, col q = x
  float mrow[8], lrow[8];
#pragma unroll
  for (int r = 0; r < 8; ++r) { mrow[r] = -3.0e38f; lrow[r] = 0.f; }

  float objr = 0.f;
#if USE_ASYNC_LDS
  auto fill_async = [&](int p, int tile) {
    async_tile_64x64(Kbase + (size_t)tile * 64 * kHd, kHd, Ks[p], threadIdx.x);
    async_tile_64x64(Vtbase + (size_t)tile * 64, kNkv, Vts[p], threadIdx.x);
  };
  fill_async(0, 0);
  if (threadIdx.x < 64) objr = objB[threadIdx.x];
#else
  uint4 kr[4], vr[4];
  auto load_kv_regs = [&](int tile) {
    const uint4* ks = (const uint4*)(Kbase + (size_t)tile * 64 * kHd);
#pragma unroll
    for (int it = 0; it < 4; ++it) kr[it] = ks[threadIdx.x + it * 128];
#pragma unroll
    for (int it = 0; it < 4; ++it) {
      int i = threadIdx.x + it * 128;
      int r = i >> 3, c = i & 7;
      vr[it] = *(const uint4*)(Vtbase + ((size_t)r << 12) + tile * 64 + c * 8);
    }
    if (threadIdx.x < 64) objr = objB[tile * 64 + threadIdx.x];
  };
  load_kv_regs(0);
#endif

  int p = 0;
  for (int tile = 0; tile < kNkv / 64; ++tile) {
#if USE_ASYNC_LDS
    __builtin_amdgcn_s_wait_asynccnt(0);
    if (threadIdx.x < 64) objs[p][threadIdx.x] = objr;
    __syncthreads();
    if (tile + 1 < kNkv / 64) {
      fill_async(p ^ 1, tile + 1);
      if (threadIdx.x < 64) objr = objB[(tile + 1) * 64 + threadIdx.x];
    }
#else
#pragma unroll
    for (int it = 0; it < 4; ++it) {
      int i = threadIdx.x + it * 128;
      int r = i >> 3, c = i & 7;
      *(uint4*)&Ks[p][r][c * 8] = kr[it];
      *(uint4*)&Vts[p][r][c * 8] = vr[it];
    }
    if (threadIdx.x < 64) objs[p][threadIdx.x] = objr;
    if (tile + 1 < kNkv / 64) load_kv_regs(tile + 1);
    __syncthreads();
#endif

    // S = Q @ K^T
    v8f s[4];
#pragma unroll
    for (int t = 0; t < 4; ++t) {
      v16h bk0 = load_bt_frag(&Ks[p][0][0], 88, 0, t * 16, lane);
      v16h bk1 = load_bt_frag(&Ks[p][0][0], 88, 32, t * 16, lane);
      v8f z = {};
      z = wmma16x16x32(qa0, bk0, z);
      s[t] = wmma16x16x32(qa1, bk1, z);
    }

    // scale * objectness, online softmax (row stats per r, reduce over lanes)
    float nm[8];
#pragma unroll
    for (int r = 0; r < 8; ++r) nm[r] = mrow[r];
#pragma unroll
    for (int t = 0; t < 4; ++t) {
      float w = kScale * objs[p][t * 16 + x];
#pragma unroll
      for (int r = 0; r < 8; ++r) {
        float sv = s[t][r] * w;
        s[t][r] = sv;
        nm[r] = fmaxf(nm[r], sv);
      }
    }
#pragma unroll
    for (int r = 0; r < 8; ++r) {
#pragma unroll
      for (int mk = 1; mk < 16; mk <<= 1)
        nm[r] = fmaxf(nm[r], __shfl_xor(nm[r], mk, 32));
    }
    float rsum[8];
#pragma unroll
    for (int r = 0; r < 8; ++r) rsum[r] = 0.f;
#pragma unroll
    for (int t = 0; t < 4; ++t)
#pragma unroll
      for (int r = 0; r < 8; ++r) {
        float pe = __expf(s[t][r] - nm[r]);
        s[t][r] = pe;
        rsum[r] += pe;
      }
    float alpha[8];
#pragma unroll
    for (int r = 0; r < 8; ++r) {
#pragma unroll
      for (int mk = 1; mk < 16; mk <<= 1)
        rsum[r] += __shfl_xor(rsum[r], mk, 32);
      alpha[r] = __expf(mrow[r] - nm[r]);
      lrow[r] = lrow[r] * alpha[r] + rsum[r];
      mrow[r] = nm[r];
    }
    if (x == 0) {
#pragma unroll
      for (int r = 0; r < 8; ++r) Lalpha[wave][g * 8 + r] = alpha[r];
    }

    // P (C-layout) -> row-major Ps[q][kv]
#pragma unroll
    for (int t = 0; t < 4; ++t)
#pragma unroll
      for (int r = 0; r < 8; ++r)
        PsW[g * 8 + r][t * 16 + x] = (_Float16)s[t][r];
    __syncthreads();

    float aq = Lalpha[wave][x];
#pragma unroll
    for (int t = 0; t < 4; ++t)
#pragma unroll
      for (int r = 0; r < 8; ++r) acc[t][r] *= aq;

    // O^T += V^T @ P^T
    v16h bp0 = load_bt_frag(&PsW[0][0], 88, 0, 0, lane);
    v16h bp1 = load_bt_frag(&PsW[0][0], 88, 32, 0, lane);
#pragma unroll
    for (int t = 0; t < 4; ++t) {
      v16h av0 = load_a_frag(&Vts[p][0][0], 88, t * 16, 0, lane);
      v16h av1 = load_a_frag(&Vts[p][0][0], 88, t * 16, 32, lane);
      acc[t] = wmma16x16x32(av0, bp0, acc[t]);
      acc[t] = wmma16x16x32(av1, bp1, acc[t]);
    }
    p ^= 1;
  }

  if (x == 0) {
#pragma unroll
    for (int r = 0; r < 8; ++r) Lsum[wave][g * 8 + r] = lrow[r];
  }
  __syncthreads();
  float linv = 1.f / Lsum[wave][x];
  int q = q0 + wave * 16 + x;
  _Float16* orow = O + ((size_t)b * kNq + q) * kDim + h * kHd;
#pragma unroll
  for (int t = 0; t < 4; ++t) {
    union { _Float16 h8[8]; uint4 v; } pk;
#pragma unroll
    for (int r = 0; r < 8; ++r) pk.h8[r] = (_Float16)(acc[t][r] * linv);
    *(uint4*)(orow + t * 16 + g * 8) = pk.v;
  }
}

// ---------------------------------------------------------------------------
// out (fp32) = A (f16 [M,1024]) @ Wo + bo, Wo pre-transposed [n][k].
// Both operands f16 -> both tiles async (or register-pipelined fallback).
// ---------------------------------------------------------------------------
__global__ __launch_bounds__(128) void oproj_kernel(
    const _Float16* __restrict__ A, const _Float16* __restrict__ Wt,
    const float* __restrict__ bias, float* __restrict__ Y) {
  __shared__ _Float16 Xs[2][64][40];
  __shared__ _Float16 Wts[2][64][40];
  const int lane = threadIdx.x & 31;
  const int wave = threadIdx.x >> 5;
  const int m0 = blockIdx.x * 64;
  const int n0 = blockIdx.y * 64;

  v8f acc[4] = {};
#if USE_ASYNC_LDS
  auto fill = [&](int p, int k0) {
    async_tile_64x32(A + (size_t)m0 * kDim + k0, kDim, Xs[p], threadIdx.x);
    async_tile_64x32(Wt + (size_t)n0 * kDim + k0, kDim, Wts[p], threadIdx.x);
  };
  fill(0, 0);
#else
  uint4 ar[2], wr[2];
  auto load_regs = [&](int k0) {
#pragma unroll
    for (int it = 0; it < 2; ++it) {
      int i = threadIdx.x + it * 128;
      int r = i >> 2, c = i & 3;
      ar[it] = ((const uint4*)(A + (size_t)(m0 + r) * kDim + k0))[c];
      wr[it] = ((const uint4*)(Wt + (size_t)(n0 + r) * kDim + k0))[c];
    }
  };
  load_regs(0);
#endif

  int p = 0;
  for (int k0 = 0; k0 < kDim; k0 += 32) {
#if USE_ASYNC_LDS
    __builtin_amdgcn_s_wait_asynccnt(0);
    __syncthreads();
    if (k0 + 32 < kDim) fill(p ^ 1, k0 + 32);
#else
#pragma unroll
    for (int it = 0; it < 2; ++it) {
      int i = threadIdx.x + it * 128;
      int r = i >> 2, c = i & 3;
      *(uint4*)&Xs[p][r][c * 8] = ar[it];
      *(uint4*)&Wts[p][r][c * 8] = wr[it];
    }
    if (k0 + 32 < kDim) load_regs(k0 + 32);
    __syncthreads();
#endif
    v16h a = load_a_frag(&Xs[p][0][0], 40, wave * 16, 0, lane);
#pragma unroll
    for (int t = 0; t < 4; ++t) {
      v16h b = load_bt_frag(&Wts[p][0][0], 40, 0, t * 16, lane);
      acc[t] = wmma16x16x32(a, b, acc[t]);
    }
    p ^= 1;
  }

#pragma unroll
  for (int t = 0; t < 4; ++t) {
    int n = n0 + t * 16 + (lane & 15);
    float bv = bias[n];
#pragma unroll
    for (int r = 0; r < 8; ++r) {
      int m = m0 + wave * 16 + (lane >> 4) * 8 + r;
      Y[(size_t)m * kDim + n] = acc[t][r] + bv;
    }
  }
}

// ---------------------------------------------------------------------------
extern "C" void kernel_launch(void* const* d_in, const int* in_sizes, int n_in,
                              void* d_out, int out_size, void* d_ws,
                              size_t ws_size, hipStream_t stream) {
  (void)in_sizes; (void)n_in; (void)out_size; (void)ws_size;
  const float* query = (const float*)d_in[0];
  const float* key   = (const float*)d_in[1];
  const float* value = (const float*)d_in[2];
  const float* obj   = (const float*)d_in[3];
  const float* Wq = (const float*)d_in[4];
  const float* bq = (const float*)d_in[5];
  const float* Wk = (const float*)d_in[6];
  const float* bk = (const float*)d_in[7];
  const float* Wv = (const float*)d_in[8];
  const float* bv = (const float*)d_in[9];
  const float* Wo = (const float*)d_in[10];
  const float* bo = (const float*)d_in[11];
  float* out = (float*)d_out;

  char* ws = (char*)d_ws;
  size_t off = 0;
  _Float16* Qh  = (_Float16*)(ws + off); off += (size_t)kBatch * kHeads * kNq  * kHd * 2;
  _Float16* Kh  = (_Float16*)(ws + off); off += (size_t)kBatch * kHeads * kNkv * kHd * 2;
  _Float16* Vth = (_Float16*)(ws + off); off += (size_t)kBatch * kHeads * kNkv * kHd * 2;
  _Float16* Ah  = (_Float16*)(ws + off); off += (size_t)kBatch * kNq * kDim * 2;
  _Float16* Wtq = (_Float16*)(ws + off); off += (size_t)kDim * kDim * 2;
  _Float16* Wtk = (_Float16*)(ws + off); off += (size_t)kDim * kDim * 2;
  _Float16* Wtv = (_Float16*)(ws + off); off += (size_t)kDim * kDim * 2;
  _Float16* Wto = (_Float16*)(ws + off); off += (size_t)kDim * kDim * 2;

  dim3 tgrid(kDim / 32, kDim / 32), tblk(32, 8);
  cvt_transpose_kernel<<<tgrid, tblk, 0, stream>>>(Wq, Wtq);
  cvt_transpose_kernel<<<tgrid, tblk, 0, stream>>>(Wk, Wtk);
  cvt_transpose_kernel<<<tgrid, tblk, 0, stream>>>(Wv, Wtv);
  cvt_transpose_kernel<<<tgrid, tblk, 0, stream>>>(Wo, Wto);

  proj_qkv_kernel<<<dim3((kBatch * kNq) / 64, kDim / 64), 128, 0, stream>>>(
      query, Wtq, bq, Qh, 10, 0);
  proj_qkv_kernel<<<dim3((kBatch * kNkv) / 64, kDim / 64), 128, 0, stream>>>(
      key, Wtk, bk, Kh, 12, 0);
  proj_qkv_kernel<<<dim3((kBatch * kNkv) / 64, kDim / 64), 128, 0, stream>>>(
      value, Wtv, bv, Vth, 12, 1);

  flash_attn_kernel<<<dim3(kBatch * kHeads, kNq / 64), 128, 0, stream>>>(
      Qh, Kh, Vth, obj, Ah);

  oproj_kernel<<<dim3((kBatch * kNq) / 64, kDim / 64), 128, 0, stream>>>(
      Ah, Wto, bo, out);
}
